// MultiHeadAttentionSAGELayer_18056042512803
// MI455X (gfx1250) — matrix operand order repaired and checked by reading, not compiled, not run on hardware
//
#include <hip/hip_runtime.h>
#include <math.h>

// ---------------------------------------------------------------------------
// MultiHeadAttentionSAGELayer for MI455X (gfx1250, wave32, WMMA bf16)
// N=100000 nodes, E=1600000 edges, 128 in / 64 edge / 128 out, 4 heads x 32
// Weights pre-permuted to WMMA B-fragment order so LDS loads are ds_load_b128.
// ---------------------------------------------------------------------------

typedef __bf16 bf16;
typedef bf16  v16bf __attribute__((ext_vector_type(16)));
typedef bf16  v8bf  __attribute__((ext_vector_type(8)));
typedef bf16  v2bf  __attribute__((ext_vector_type(2)));
typedef float v8f   __attribute__((ext_vector_type(8)));

#define NDIM  128
#define EDIM  64
#define KDIM  192   // NDIM + EDIM
#define NHEAD 4

__device__ __forceinline__ v8f wmma_bf16(v16bf a, v16bf b, v8f c) {
  // D = A(16x32 bf16) x B(32x16 bf16) + C(16x16 f32)
  return __builtin_amdgcn_wmma_f32_16x16x32_bf16(
      /*neg_a=*/false, a, /*neg_b=*/false, b,
      /*c_mod=*/(short)0, c, /*reuse_a=*/false, /*reuse_b=*/false);
}

__device__ __forceinline__ v16bf fuse16(v8bf lo, v8bf hi) {
  return __builtin_shufflevector(lo, hi, 0, 1, 2, 3, 4, 5, 6, 7,
                                 8, 9, 10, 11, 12, 13, 14, 15);
}

__device__ __forceinline__ void atomic_add_f(float* p, float v) {
  __hip_atomic_fetch_add(p, v, __ATOMIC_RELAXED, __HIP_MEMORY_SCOPE_AGENT);
}

// float atomic max via int/uint punning (works with -inf init, mixed signs)
__device__ __forceinline__ void atomic_max_f(float* p, float v) {
  if (v >= 0.0f)
    __hip_atomic_fetch_max((int*)p, __float_as_int(v),
                           __ATOMIC_RELAXED, __HIP_MEMORY_SCOPE_AGENT);
  else
    __hip_atomic_fetch_min((unsigned int*)p, __float_as_uint(v),
                           __ATOMIC_RELAXED, __HIP_MEMORY_SCOPE_AGENT);
}

// ---------------------------------------------------------------------------
// Kernel 0: zero h_neigh accumulator (d_out), denom; emax = -inf
// ---------------------------------------------------------------------------
__global__ void init_kernel(float* __restrict__ acc, float* __restrict__ emax,
                            float* __restrict__ denom, int N) {
  long stride = (long)gridDim.x * blockDim.x;
  long i0 = (long)blockIdx.x * blockDim.x + threadIdx.x;
  long total = (long)N * NDIM;
  for (long i = i0; i < total; i += stride) acc[i] = 0.0f;
  long th = (long)N * NHEAD;
  for (long i = i0; i < th; i += stride) { emax[i] = -INFINITY; denom[i] = 0.0f; }
}

// ---------------------------------------------------------------------------
// Kernel 1: weight prep into WMMA B-fragment order:
//   frag idx = ((jt*KC + kc)*32 + lane)*16 + i
//   holds B[k][n] with n = jt*16 + (lane&15), k = kc*32 + (lane>>4)*16 + i
// Also V[k,h] = sum_d W_proj[h*32+d, k]*attn[h,d];  c0[h] = bias_h . attn_h.
// ---------------------------------------------------------------------------
__global__ void prep_kernel(const float* __restrict__ Wp, const float* __restrict__ Wpb,
                            const float* __restrict__ av, const float* __restrict__ Wo,
                            bf16* __restrict__ wpB, bf16* __restrict__ woB,
                            float* __restrict__ V, float* __restrict__ c0) {
  int tid = threadIdx.x;
  // W_proj fragments: 8 jt * 6 kc * 32 lanes * 16 = 24576
  for (int idx = tid; idx < 8 * 6 * 32 * 16; idx += 256) {
    int i = idx & 15;
    int ln = (idx >> 4) & 31;
    int rest = idx >> 9;            // jt*6 + kc
    int kc = rest % 6, jt = rest / 6;
    int n = jt * 16 + (ln & 15);
    int k = kc * 32 + (ln >> 4) * 16 + i;
    wpB[idx] = (bf16)Wp[n * KDIM + k];
  }
  // W_out fragments: 8 jt * 4 kc * 32 lanes * 16 = 16384
  for (int idx = tid; idx < 8 * 4 * 32 * 16; idx += 256) {
    int i = idx & 15;
    int ln = (idx >> 4) & 31;
    int rest = idx >> 9;            // jt*4 + kc
    int kc = rest & 3, jt = rest >> 2;
    int n = jt * 16 + (ln & 15);
    int k = kc * 32 + (ln >> 4) * 16 + i;
    woB[idx] = (bf16)Wo[n * NDIM + k];
  }
  for (int idx = tid; idx < KDIM * NHEAD; idx += 256) {
    int k = idx >> 2, h = idx & 3;
    float s = 0.0f;
    for (int d = 0; d < 32; ++d) s += Wp[(h * 32 + d) * KDIM + k] * av[h * 32 + d];
    V[idx] = s;
  }
  if (tid < NHEAD) {
    float s = 0.0f;
    for (int d = 0; d < 32; ++d) s += Wpb[tid * 32 + d] * av[tid * 32 + d];
    c0[tid] = s;
  }
}

// ---------------------------------------------------------------------------
// Kernel 2: per-edge attention logits e[E,4] = X . V + c0 ; segment max.
// One wave per edge (grid-stride).
// ---------------------------------------------------------------------------
__global__ __launch_bounds__(256) void edge_score_kernel(
    const float* __restrict__ nfeats, const float* __restrict__ efeats,
    const int* __restrict__ src, const int* __restrict__ dst,
    const float* __restrict__ V, const float* __restrict__ c0,
    float* __restrict__ esc, float* __restrict__ emax, int E) {
  int lane = threadIdx.x & 31;
  long wid = (long)blockIdx.x * 8 + (threadIdx.x >> 5);
  long nwaves = (long)gridDim.x * 8;
  for (long e = wid; e < E; e += nwaves) {
    int s = src[e];
    float a0 = 0.f, a1 = 0.f, a2 = 0.f, a3 = 0.f;
    #pragma unroll
    for (int k = lane; k < KDIM; k += 32) {
      float x = (k < NDIM) ? nfeats[(size_t)s * NDIM + k]
                           : efeats[(size_t)e * EDIM + (k - NDIM)];
      a0 += x * V[k * 4 + 0];
      a1 += x * V[k * 4 + 1];
      a2 += x * V[k * 4 + 2];
      a3 += x * V[k * 4 + 3];
    }
    #pragma unroll
    for (int off = 16; off > 0; off >>= 1) {
      a0 += __shfl_down(a0, off, 32);
      a1 += __shfl_down(a1, off, 32);
      a2 += __shfl_down(a2, off, 32);
      a3 += __shfl_down(a3, off, 32);
    }
    if (lane == 0) {
      int d = dst[e];
      float v0 = a0 + c0[0], v1 = a1 + c0[1], v2 = a2 + c0[2], v3 = a3 + c0[3];
      esc[e * 4 + 0] = v0; esc[e * 4 + 1] = v1;
      esc[e * 4 + 2] = v2; esc[e * 4 + 3] = v3;
      atomic_max_f(&emax[(size_t)d * 4 + 0], v0);
      atomic_max_f(&emax[(size_t)d * 4 + 1], v1);
      atomic_max_f(&emax[(size_t)d * 4 + 2], v2);
      atomic_max_f(&emax[(size_t)d * 4 + 3], v3);
    }
  }
}

// ---------------------------------------------------------------------------
// Kernel 3: edge aggregate. Each wave owns a 16-edge tile: gather X into LDS
// (bf16), x_proj = X @ W_proj^T via 48 WMMAs, multiply by ex=exp(e-emax[dst]),
// scatter-add to acc[dst] (d_out), accumulate denom[dst].
// ---------------------------------------------------------------------------
__global__ __launch_bounds__(256) void edge_aggregate_kernel(
    const float* __restrict__ nfeats, const float* __restrict__ efeats,
    const int* __restrict__ src, const int* __restrict__ dst,
    const bf16* __restrict__ wpB, const float* __restrict__ Wpb,
    const float* __restrict__ esc, const float* __restrict__ emax,
    float* __restrict__ denom, float* __restrict__ acc, int E) {
  __shared__ __attribute__((aligned(32))) bf16 sW[8 * 6 * 32 * 16]; // 49152 B frag order
  __shared__ __attribute__((aligned(16))) bf16 sX[8][16][KDIM];     // 49152 B A tiles
  __shared__ float sEx[8][16][NHEAD];                               //  2048 B
  __shared__ int   sDst[8][16];                                     //   512 B

  const int tid = threadIdx.x;
  const int w = tid >> 5, lane = tid & 31;
  const int m = lane & 15, hi = lane >> 4;

  // stage W_proj fragments block-wide (flat copy, already permuted)
  for (int i = tid; i < 8 * 6 * 32 * 16; i += 256) sW[i] = wpB[i];

  const long ebase = (long)blockIdx.x * 128 + (long)w * 16;

  // gather 16 rows of X = [nfeats[src[e]] | efeats[e]] into LDS as bf16 pairs
  for (int r = 0; r < 16; ++r) {
    long e = ebase + r;
    bool ok = e < E;
    int s = ok ? src[e] : 0;
    #pragma unroll
    for (int kp = lane; kp < KDIM / 2; kp += 32) {   // 3 iters of paired loads
      int k = kp * 2;
      float f0 = 0.0f, f1 = 0.0f;
      if (ok) {
        if (k < NDIM) {
          f0 = nfeats[(size_t)s * NDIM + k];
          f1 = nfeats[(size_t)s * NDIM + k + 1];
        } else {
          f0 = efeats[(size_t)e * EDIM + (k - NDIM)];
          f1 = efeats[(size_t)e * EDIM + (k - NDIM) + 1];
        }
      }
      v2bf p = {(bf16)f0, (bf16)f1};
      *(v2bf*)&sX[w][r][k] = p;                      // 4B LDS store
    }
  }

  // softmax weights ex = exp(e - emax[dst]); accumulate denom
  if (lane < 16) {
    long e = ebase + lane;
    if (e < E) {
      int d = dst[e];
      sDst[w][lane] = d;
      #pragma unroll
      for (int h = 0; h < NHEAD; ++h) {
        float ex = __expf(esc[e * 4 + h] - emax[(size_t)d * 4 + h]);
        sEx[w][lane][h] = ex;
        atomic_add_f(&denom[(size_t)d * 4 + h], ex);
      }
    } else {
      sDst[w][lane] = 0;
      #pragma unroll
      for (int h = 0; h < NHEAD; ++h) sEx[w][lane][h] = 0.0f;
    }
  }
  __syncthreads();

  // A fragments (contiguous k order, matching B frags): two 8-elem runs
  // at k = kc*32 + hi*16 and k = kc*32 + hi*16 + 8  -> 2x ds_load_b128
  v16bf A[6];
  {
    const v8bf* row = (const v8bf*)&sX[w][m][0];     // 24 chunks of 8 bf16
    #pragma unroll
    for (int kc = 0; kc < 6; ++kc)
      A[kc] = fuse16(row[kc * 4 + 2 * hi], row[kc * 4 + 2 * hi + 1]);
  }

  // 8 output tiles of 16 columns; D row = r + 8*hi, col = m
  #pragma unroll
  for (int jt = 0; jt < 8; ++jt) {
    int j = jt * 16 + m;
    v8f cacc = {0.f, 0.f, 0.f, 0.f, 0.f, 0.f, 0.f, 0.f};
    #pragma unroll
    for (int kc = 0; kc < 6; ++kc) {
      v16bf B = *(const v16bf*)&sW[((jt * 6 + kc) * 32 + lane) * 16]; // 32B aligned
      cacc = wmma_bf16(A[kc], B, cacc);
    }
    float bias = Wpb[j];
    int head = jt >> 1;
    #pragma unroll
    for (int r = 0; r < 8; ++r) {
      int mr = r + hi * 8;
      float val = (cacc[r] + bias) * sEx[w][mr][head];
      atomic_add_f(&acc[(size_t)sDst[w][mr] * NDIM + j], val);
    }
  }
}

// ---------------------------------------------------------------------------
// Kernel 4: node output. h_in = acc/denom + nfeats (bf16) -> WMMA with W_out
// -> bias+ReLU -> LayerNorm (in-register row stats via shfl_xor) -> d_out.
// d_out is read (acc) then overwritten; rows are block/wave exclusive.
// ---------------------------------------------------------------------------
__global__ __launch_bounds__(256) void node_out_kernel(
    const float* __restrict__ nfeats, const bf16* __restrict__ woB,
    const float* __restrict__ Wob, const float* __restrict__ denom,
    const float* __restrict__ lng, const float* __restrict__ lnb,
    float* out, int N) {
  __shared__ __attribute__((aligned(32))) bf16 sW[8 * 4 * 32 * 16]; // 32768 B
  __shared__ __attribute__((aligned(16))) bf16 sA[8][16][NDIM];     // 32768 B

  const int tid = threadIdx.x;
  const int w = tid >> 5, lane = tid & 31;
  const int m = lane & 15, hi = lane >> 4;

  for (int i = tid; i < 8 * 4 * 32 * 16; i += 256) sW[i] = woB[i];

  const long nbase = (long)blockIdx.x * 128 + (long)w * 16;

  for (int r = 0; r < 16; ++r) {
    long n = nbase + r;
    bool ok = n < N;
    #pragma unroll
    for (int kp = lane; kp < NDIM / 2; kp += 32) {   // 2 iters of paired loads
      int k = kp * 2;
      float f0 = 0.0f, f1 = 0.0f;
      if (ok) {
        float dn = denom[(size_t)n * 4 + (k >> 5)];  // pair shares a head
        float inv = (dn > 0.0f) ? (1.0f / dn) : 0.0f;
        f0 = out[(size_t)n * NDIM + k] * inv + nfeats[(size_t)n * NDIM + k];
        f1 = out[(size_t)n * NDIM + k + 1] * inv + nfeats[(size_t)n * NDIM + k + 1];
      }
      v2bf p = {(bf16)f0, (bf16)f1};
      *(v2bf*)&sA[w][r][k] = p;
    }
  }
  __syncthreads();

  v16bf A[4];
  {
    const v8bf* row = (const v8bf*)&sA[w][m][0];     // 16 chunks of 8 bf16
    #pragma unroll
    for (int kc = 0; kc < 4; ++kc)
      A[kc] = fuse16(row[kc * 4 + 2 * hi], row[kc * 4 + 2 * hi + 1]);
  }

  v8f ct[8];
  #pragma unroll
  for (int jt = 0; jt < 8; ++jt) {
    int j = jt * 16 + m;
    v8f cacc = {0.f, 0.f, 0.f, 0.f, 0.f, 0.f, 0.f, 0.f};
    #pragma unroll
    for (int kc = 0; kc < 4; ++kc) {
      v16bf B = *(const v16bf*)&sW[((jt * 4 + kc) * 32 + lane) * 16];
      cacc = wmma_bf16(A[kc], B, cacc);
    }
    float bias = Wob[j];
    #pragma unroll
    for (int r = 0; r < 8; ++r) {
      float v = cacc[r] + bias;
      cacc[r] = v > 0.0f ? v : 0.0f;   // ReLU
    }
    ct[jt] = cacc;
  }

  // row stats: sum/sumsq over 128 cols = over 8 jt tiles and 16 lanes/N-group
  float sum[8], sq[8];
  #pragma unroll
  for (int r = 0; r < 8; ++r) { sum[r] = 0.0f; sq[r] = 0.0f; }
  #pragma unroll
  for (int jt = 0; jt < 8; ++jt)
    #pragma unroll
    for (int r = 0; r < 8; ++r) {
      float v = ct[jt][r];
      sum[r] += v; sq[r] += v * v;
    }
  #pragma unroll
  for (int off = 1; off < 16; off <<= 1)
    #pragma unroll
    for (int r = 0; r < 8; ++r) {
      sum[r] += __shfl_xor(sum[r], off, 32);
      sq[r]  += __shfl_xor(sq[r],  off, 32);
    }

  float g8[8], b8[8];
  #pragma unroll
  for (int jt = 0; jt < 8; ++jt) {
    int j = jt * 16 + m;
    g8[jt] = lng[j];
    b8[jt] = lnb[j];
  }

  #pragma unroll
  for (int r = 0; r < 8; ++r) {
    long n = nbase + r + hi * 8;
    if (n < N) {
      float mu = sum[r] * (1.0f / 128.0f);
      float var = sq[r] * (1.0f / 128.0f) - mu * mu;
      float inv = rsqrtf(var + 1e-5f);
      #pragma unroll
      for (int jt = 0; jt < 8; ++jt) {
        int j = jt * 16 + m;
        out[(size_t)n * NDIM + j] = (ct[jt][r] - mu) * inv * g8[jt] + b8[jt];
      }
    }
  }
}

// ---------------------------------------------------------------------------
extern "C" void kernel_launch(void* const* d_in, const int* in_sizes, int n_in,
                              void* d_out, int out_size, void* d_ws, size_t ws_size,
                              hipStream_t stream) {
  const float* nfeats = (const float*)d_in[0];
  const float* efeats = (const float*)d_in[1];
  const int*   src    = (const int*)d_in[2];
  const int*   dst    = (const int*)d_in[3];
  const float* Wp     = (const float*)d_in[4];
  const float* Wpb    = (const float*)d_in[5];
  const float* av     = (const float*)d_in[6];
  const float* Wo     = (const float*)d_in[7];
  const float* Wob    = (const float*)d_in[8];
  const float* lng    = (const float*)d_in[9];
  const float* lnb    = (const float*)d_in[10];

  const int N = in_sizes[0] / NDIM;   // 100000
  const int E = in_sizes[1] / EDIM;   // 1600000

  // workspace layout (~29 MB)
  char* p = (char*)d_ws;
  bf16*  wpB   = (bf16*)p;  p += (size_t)8 * 6 * 32 * 16 * sizeof(bf16);
  bf16*  woB   = (bf16*)p;  p += (size_t)8 * 4 * 32 * 16 * sizeof(bf16);
  float* V     = (float*)p; p += (size_t)KDIM * NHEAD * sizeof(float);
  float* c0    = (float*)p; p += (size_t)NHEAD * sizeof(float);
  float* esc   = (float*)p; p += (size_t)E * NHEAD * sizeof(float);
  float* emax  = (float*)p; p += (size_t)N * NHEAD * sizeof(float);
  float* denom = (float*)p; p += (size_t)N * NHEAD * sizeof(float);

  float* acc = (float*)d_out;  // h_neigh accumulator, later final output

  init_kernel<<<4096, 256, 0, stream>>>(acc, emax, denom, N);
  prep_kernel<<<1, 256, 0, stream>>>(Wp, Wpb, av, Wo, wpB, woB, V, c0);
  edge_score_kernel<<<4096, 256, 0, stream>>>(nfeats, efeats, src, dst, V, c0,
                                              esc, emax, E);
  edge_aggregate_kernel<<<(E + 127) / 128, 256, 0, stream>>>(
      nfeats, efeats, src, dst, wpB, Wpb, esc, emax, denom, acc, E);
  node_out_kernel<<<(N + 127) / 128, 256, 0, stream>>>(
      nfeats, woB, Wob, denom, lng, lnb, acc, N);
}
